// TuckER_55078660604039
// MI455X (gfx1250) — compile-verified, matrix operand order checked by reading, not run on hardware
//
#include <hip/hip_runtime.h>
#include <hip/hip_bf16.h>

typedef __attribute__((ext_vector_type(16))) _Float16 v16h;
typedef __attribute__((ext_vector_type(8)))  float    v8f;

union V16 { v16h h; unsigned u[8]; };

#define NENT   100000
#define D      256
#define BATCH  512
#define NP1    32896   // 256*257/2
#define NP2    32640   // 256*255/2
#define NCHUNK 4       // i-chunks for the coefficient GEMM (deterministic reduce)

// ---------------- coefficient build: C_sym[11][256], C_asym[11][256] ----------------
__global__ void k_coeff(const float* __restrict__ R1, const float* __restrict__ R2,
                        const float* __restrict__ R3, float* __restrict__ cs,
                        float* __restrict__ ca) {
  int i = threadIdx.x;  // 256 threads
  for (int rel = 0; rel < 11; ++rel) {
    float s, a;
    if (rel < 3)      { s = R1[rel * 256 + i];               a = 0.f; }
    else if (rel < 8) { s = 0.f;                             a = R2[(rel - 3) * 256 + i]; }
    else              { s = R3[(rel - 8) * 512 + i];         a = R3[(rel - 8) * 512 + 256 + i]; }
    cs[rel * 256 + i] = s;
    ca[rel * 256 + i] = a;
  }
}

// ---------------- P_part[chunk][rel][m] = sum_{i in chunk} C[rel][i] * W[i][m] ----------------
__global__ __launch_bounds__(256) void k_pgemm(const float* __restrict__ W,
                                               const float* __restrict__ coeff,
                                               float* __restrict__ part, int np) {
  __shared__ float c[11 * 256];
  for (int i = threadIdx.x; i < 11 * 256; i += 256) c[i] = coeff[i];
  __syncthreads();
  int m = blockIdx.x * 256 + threadIdx.x;
  int chunk = blockIdx.y;
  if (m >= np) return;
  float acc[11];
#pragma unroll
  for (int r = 0; r < 11; ++r) acc[r] = 0.f;
  int ibase = chunk * (256 / NCHUNK);
#pragma unroll 4
  for (int ii = 0; ii < 256 / NCHUNK; ++ii) {
    float wv = W[(size_t)(ibase + ii) * np + m];
#pragma unroll
    for (int r = 0; r < 11; ++r) acc[r] += c[r * 256 + ibase + ii] * wv;
  }
#pragma unroll
  for (int r = 0; r < 11; ++r) part[((size_t)chunk * 11 + r) * np + m] = acc[r];
}

__global__ void k_reduce(const float* __restrict__ part, float* __restrict__ p, int np) {
  int t = blockIdx.x * 256 + threadIdx.x;
  if (t >= 11 * np) return;
  int rel = t / np, m = t - rel * np;
  float s = 0.f;
#pragma unroll
  for (int ch = 0; ch < NCHUNK; ++ch) s += part[((size_t)ch * 11 + rel) * np + m];
  p[t] = s;
}

// ---------------- M[rel][j][k] = sym(p1[rel]) + asym(p2[rel]) ----------------
__global__ void k_M(const float* __restrict__ p1, const float* __restrict__ p2,
                    float* __restrict__ M) {
  int t = blockIdx.x * 256 + threadIdx.x;  // 11*65536 exactly
  int rel = t >> 16;
  int jk = t & 65535;
  int j = jk >> 8, k = jk & 255;
  const int n2 = 512;  // 2*n
  float s;
  if (j > k)      s = p1[rel * NP1 + (n2 - k + 1) * k / 2 + (j - k)];
  else if (j < k) s = p1[rel * NP1 + (n2 - j + 1) * j / 2 + (k - j)];
  else            s = 2.f * p1[rel * NP1 + (n2 - j + 1) * j / 2];
  float a = 0.f;
  if (k > j)      a =  p2[rel * NP2 + j * (n2 - j - 1) / 2 + (k - j - 1)];
  else if (k < j) a = -p2[rel * NP2 + k * (n2 - k - 1) / 2 + (j - k - 1)];
  M[t] = s + a;
}

// ---------------- batchnorm stats over gathered E rows (per feature) ----------------
__global__ void k_bn0(const int* __restrict__ e1, const float* __restrict__ E,
                      float* __restrict__ stats) {
  int d = threadIdx.x;  // 256
  float sum = 0.f, sq = 0.f;
  for (int b = 0; b < BATCH; ++b) {
    float v = E[(size_t)e1[b] * D + d];
    sum += v; sq += v * v;
  }
  float mu = sum * (1.f / BATCH);
  float var = sq * (1.f / BATCH) - mu * mu;
  stats[d] = mu;
  stats[256 + d] = rsqrtf(var + 1e-5f);
}

__global__ void k_x(const int* __restrict__ e1, const float* __restrict__ E,
                    const float* __restrict__ stats, const float* __restrict__ g0,
                    const float* __restrict__ b0p, float* __restrict__ x) {
  int b = blockIdx.x, d = threadIdx.x;
  float v = E[(size_t)e1[b] * D + d];
  x[b * D + d] = (v - stats[d]) * stats[256 + d] * g0[d] + b0p[d];
}

// ---------------- x2[b,k] = sum_j x[b,j] * M[rel_b][j][k] ----------------
__global__ void k_x2(const int* __restrict__ r_idx, const float* __restrict__ x,
                     const float* __restrict__ M, float* __restrict__ x2) {
  __shared__ float xr[D];
  int b = blockIdx.x, k = threadIdx.x;
  xr[k] = x[b * D + k];
  __syncthreads();
  // REL_PERM = [8,0,3,9,4,5,6,7,1,10,2] packed as nibbles
  const unsigned long long P = 0x2A176549308ULL;
  int rr = (int)((P >> (4 * r_idx[b])) & 0xF);
  const float* Mb = M + (size_t)rr * 65536;
  float acc = 0.f;
#pragma unroll 4
  for (int j = 0; j < D; ++j) acc += xr[j] * Mb[j * D + k];
  x2[b * D + k] = acc;
}

__global__ void k_bn1(const float* __restrict__ x2, float* __restrict__ stats) {
  int d = threadIdx.x;
  float sum = 0.f, sq = 0.f;
  for (int b = 0; b < BATCH; ++b) {
    float v = x2[b * D + d];
    sum += v; sq += v * v;
  }
  float mu = sum * (1.f / BATCH);
  float var = sq * (1.f / BATCH) - mu * mu;
  stats[512 + d] = mu;
  stats[768 + d] = rsqrtf(var + 1e-5f);
}

__global__ void k_y(const float* __restrict__ x2, const float* __restrict__ stats,
                    const float* __restrict__ g1, const float* __restrict__ b1p,
                    _Float16* __restrict__ y) {
  int b = blockIdx.x, d = threadIdx.x;
  float v = (x2[b * D + d] - stats[512 + d]) * stats[768 + d] * g1[d] + b1p[d];
  y[b * D + d] = (_Float16)v;
}

// ---------------- main WMMA GEMM: out = sigmoid(Y[512,256] @ E^T[256,100000]) ----------------
#define BT 128  // batch rows per block
#define ET 64   // entities per block
#define RS 132  // row stride in dwords (264 f16)

__global__ __launch_bounds__(256) void k_gemm(const _Float16* __restrict__ Yh,
                                              const float* __restrict__ E,
                                              float* __restrict__ out) {
  __shared__ unsigned Ylds[BT * RS];
  __shared__ unsigned Elds[ET * RS];
  const int tid = threadIdx.x;
  const int b0 = blockIdx.y * BT;
  const int e0 = blockIdx.x * ET;

  // stage Y tile (already f16) — coalesced dword copy
  const unsigned* Yg = reinterpret_cast<const unsigned*>(Yh);
  for (int idx = tid; idx < BT * 128; idx += 256) {
    int r = idx >> 7, c = idx & 127;
    Ylds[r * RS + c] = Yg[(b0 + r) * 128 + c];
  }
  // stage E tile, fp32 -> f16 pack
  for (int idx = tid; idx < ET * 128; idx += 256) {
    int r = idx >> 7, c = idx & 127;
    int e = e0 + r;
    int esrc = (e < NENT) ? e : 0;
    float f0 = E[(size_t)esrc * D + 2 * c];
    float f1 = E[(size_t)esrc * D + 2 * c + 1];
    union { _Float16 h[2]; unsigned u; } pk;
    pk.h[0] = (_Float16)f0;
    pk.h[1] = (_Float16)f1;
    Elds[r * RS + c] = pk.u;
  }
  __syncthreads();

  const int w   = tid >> 5;
  const int lane = tid & 31;
  const int hh  = (lane >> 4) & 1;
  const int l16 = lane & 15;
  const int bt0 = (w >> 1) * 2;  // {0,2,4,6} -> 8 b-subtiles as pairs
  const int et0 = (w & 1) * 2;   // {0,2}     -> 4 e-subtiles as pairs

  v8f acc[2][2] = {};
  for (int kc = 0; kc < 8; ++kc) {   // K = 256 in chunks of 32
    V16 a[2], bm[2];
#pragma unroll
    for (int t = 0; t < 2; ++t) {
      int arow = (bt0 + t) * 16 + l16;
      int brow = (et0 + t) * 16 + l16;
#pragma unroll
      for (int v = 0; v < 8; ++v) {
        // f16 A/B layout: reg v, half-group hh holds K pair 16*(v>>2)+8*hh+2*(v&3)
        int cd = kc * 16 + ((v >> 2) << 3) + (hh << 2) + (v & 3);
        a[t].u[v]  = Ylds[arow * RS + cd];
        bm[t].u[v] = Elds[brow * RS + cd];
      }
    }
#pragma unroll
    for (int i = 0; i < 2; ++i)
#pragma unroll
      for (int j = 0; j < 2; ++j)
        acc[i][j] = __builtin_amdgcn_wmma_f32_16x16x32_f16(
            false, a[i].h, false, bm[j].h, (short)0, acc[i][j], false, false);
  }

  // C/D layout: lane half hh, reg q -> row m = q + 8*hh, col n = lane&15
#pragma unroll
  for (int i = 0; i < 2; ++i) {
    int bb = b0 + (bt0 + i) * 16 + 8 * hh;
#pragma unroll
    for (int j = 0; j < 2; ++j) {
      int ee = e0 + (et0 + j) * 16 + l16;
      if (ee < NENT) {
#pragma unroll
        for (int q = 0; q < 8; ++q) {
          float val = acc[i][j][q];
          out[(size_t)(bb + q) * NENT + ee] = 1.0f / (1.0f + __expf(-val));
        }
      }
    }
  }
}

extern "C" void kernel_launch(void* const* d_in, const int* in_sizes, int n_in,
                              void* d_out, int out_size, void* d_ws, size_t ws_size,
                              hipStream_t stream) {
  const int*   e1   = (const int*)d_in[0];
  const int*   ridx = (const int*)d_in[1];
  const float* E    = (const float*)d_in[2];
  const float* R1   = (const float*)d_in[3];
  const float* R2   = (const float*)d_in[4];
  const float* R3   = (const float*)d_in[5];
  const float* W1   = (const float*)d_in[6];
  const float* W2   = (const float*)d_in[7];
  const float* g0   = (const float*)d_in[8];
  const float* b0   = (const float*)d_in[9];
  const float* g1   = (const float*)d_in[10];
  const float* b1   = (const float*)d_in[11];
  float* out = (float*)d_out;

  float* ws = (float*)d_ws;
  float* cs    = ws;                    // 2816
  float* ca    = cs + 2816;             // 2816
  float* p1    = ca + 2816;             // 11*NP1 = 361856
  float* p2    = p1 + 11 * NP1;         // 11*NP2 = 359040
  float* Mw    = p2 + 11 * NP2;         // 720896
  float* xw    = Mw + 11 * 65536;       // 131072
  float* x2w   = xw + BATCH * D;        // 131072
  float* stats = x2w + BATCH * D;       // 1024
  float* yhf   = stats + 1024;          // 65536 float slots (131072 f16)
  float* part  = yhf + 65536;           // NCHUNK*11*NP1 = 1447424
  _Float16* yh = reinterpret_cast<_Float16*>(yhf);

  k_coeff<<<1, 256, 0, stream>>>(R1, R2, R3, cs, ca);

  k_pgemm<<<dim3((NP1 + 255) / 256, NCHUNK), 256, 0, stream>>>(W1, cs, part, NP1);
  k_reduce<<<(11 * NP1 + 255) / 256, 256, 0, stream>>>(part, p1, NP1);
  k_pgemm<<<dim3((NP2 + 255) / 256, NCHUNK), 256, 0, stream>>>(W2, ca, part, NP2);
  k_reduce<<<(11 * NP2 + 255) / 256, 256, 0, stream>>>(part, p2, NP2);

  k_M<<<(11 * 65536) / 256, 256, 0, stream>>>(p1, p2, Mw);

  k_bn0<<<1, 256, 0, stream>>>(e1, E, stats);
  k_x<<<BATCH, 256, 0, stream>>>(e1, E, stats, g0, b0, xw);
  k_x2<<<BATCH, 256, 0, stream>>>(ridx, xw, Mw, x2w);
  k_bn1<<<1, 256, 0, stream>>>(x2w, stats);
  k_y<<<BATCH, 256, 0, stream>>>(x2w, stats, g1, b1, yh);

  k_gemm<<<dim3((NENT + ET - 1) / ET, BATCH / BT), 256, 0, stream>>>(yh, E, out);
}
